// VectorQuantizer_41403484733547
// MI455X (gfx1250) — compile-verified, matrix operand order checked by reading, not run on hardware
//
#include <hip/hip_runtime.h>
#include <hip/hip_bf16.h>

// ---------------------------------------------------------------------------
// VQ-VAE vector quantizer for MI455X (gfx1250, wave32, WMMA + TDM).
// B=4, S=2048, D=256, K=8192 -> N = 8192 query rows.
// Core: argmin_k ( ||c_k||^2 - 2 * x . c_k ) via v_wmma_f32_16x16x32_bf16,
// with codebook tiles DMA'd into LDS by the Tensor Data Mover (TENSORcnt),
// double-buffered, hardware-padded to a conflict-free 264-element stride.
// ---------------------------------------------------------------------------

#define VQ_D      256
#define VQ_K      8192
#define VQ_N      8192            // B*S
#define VQ_ND     (VQ_N * VQ_D)   // 2,097,152
#define VQ_BETA   0.25f
#define VQ_PARTS  8               // K-partitions (grid = 32 row-blocks x 8)
#define VQ_KPART  (VQ_K / VQ_PARTS)     // 1024 codes per partition
#define VQ_NCHUNK (VQ_KPART / 16)       // 64 chunks of 16 codes
#define LDS_STRIDE 264                  // bf16 elems per code row in LDS

typedef __attribute__((ext_vector_type(16))) __bf16        v16bf;
typedef __attribute__((ext_vector_type(4)))  __bf16        v4bf;
typedef __attribute__((ext_vector_type(8)))  float         v8f;
typedef __attribute__((ext_vector_type(4)))  unsigned int  v4u;
typedef __attribute__((ext_vector_type(4)))  int           i4;
typedef __attribute__((ext_vector_type(8)))  int           i8;

union Frag32B {      // one 16x32-K bf16 WMMA operand slice per lane (32 bytes)
    v16bf bf;
    v4u   u[2];
};

// generic -> LDS(addrspace 3) byte offset, for the TDM descriptor
static __device__ inline unsigned lds_offset(const void* p) {
    return (unsigned)(unsigned long long)
        (__attribute__((address_space(3))) const void*)p;
}

// ---------------------------------------------------------------------------
// TDM: DMA one 16 x 256 bf16 codebook tile (row-major, stride 256) into LDS,
// padding 4 DWORDs after every 128 DWORDs -> LDS row stride = 264 bf16.
// ---------------------------------------------------------------------------
static __device__ inline void tdm_load_tile(const __bf16* gsrc, unsigned lds_addr)
{
    const unsigned long long ga = (unsigned long long)gsrc;
    v4u g0;
    g0[0] = 1u;                                    // count=1, user desc
    g0[1] = lds_addr;                              // LDS byte address
    g0[2] = (unsigned)(ga & 0xFFFFFFFFu);          // global_addr[31:0]
    g0[3] = (unsigned)((ga >> 32) & 0x1FFFFFFu)    // global_addr[56:32]
          | (2u << 30);                            // type = 2 ("image")
    i8 g1;
    g1[0] = (1 << 16)                              // data_size = 2 bytes
          | (1 << 20)                              // pad_enable
          | (6 << 22)                              // pad_interval: 128 DWORDs
          | (3 << 25);                             // pad_amount:   4 DWORDs
    g1[1] = (int)(((unsigned)VQ_D & 0xFFFFu) << 16);   // tensor_dim0[15:0]
    g1[2] = (int)(((unsigned)VQ_K & 0xFFFFu) << 16);   // dim0[31:16]=0 | tensor_dim1[15:0]
    g1[3] = (int)((unsigned)VQ_D << 16);               // dim1[31:16]=0 | tile_dim0=256
    g1[4] = 16;                                        // tile_dim1=16, tile_dim2=0
    g1[5] = VQ_D;                                      // tensor_dim0_stride[31:0]
    g1[6] = (int)((unsigned)VQ_D << 16);               // stride0[47:32]=0 | dim1_stride[15:0]
    g1[7] = 0;                                         // dim1_stride[47:16]=0
    const i4 z4 = {0, 0, 0, 0};
#if __clang_major__ >= 23
    const i8 z8 = {0, 0, 0, 0, 0, 0, 0, 0};
    __builtin_amdgcn_tensor_load_to_lds(g0, g1, z4, z4, z8, 0);
#else
    __builtin_amdgcn_tensor_load_to_lds(g0, g1, z4, z4, 0);
#endif
}

// ---------------------------------------------------------------------------
// Kernel 1: convert z_e and codebook to bf16 (workspace). 4 elems / thread.
// ---------------------------------------------------------------------------
__global__ __launch_bounds__(256) void vq_prep_convert(
    const float* __restrict__ ze, const float* __restrict__ cb,
    unsigned short* __restrict__ ze_bf, unsigned short* __restrict__ cb_bf)
{
    const int t = blockIdx.x * blockDim.x + threadIdx.x;   // < VQ_ND/4
    float4 a = ((const float4*)ze)[t];
    float4 b = ((const float4*)cb)[t];
    v4bf oa, ob;
    oa[0] = (__bf16)a.x; oa[1] = (__bf16)a.y; oa[2] = (__bf16)a.z; oa[3] = (__bf16)a.w;
    ob[0] = (__bf16)b.x; ob[1] = (__bf16)b.y; ob[2] = (__bf16)b.z; ob[3] = (__bf16)b.w;
    *(v4bf*)(ze_bf + 4 * t) = oa;
    *(v4bf*)(cb_bf + 4 * t) = ob;
}

// ---------------------------------------------------------------------------
// Kernel 2: per-code squared norms; also zero the loss accumulator.
// ---------------------------------------------------------------------------
__global__ __launch_bounds__(256) void vq_prep_cnorms(
    const float* __restrict__ cb, float* __restrict__ cnorms,
    float* __restrict__ loss_slot)
{
    const int k = blockIdx.x * blockDim.x + threadIdx.x;   // < VQ_K
    const float4* c4 = (const float4*)(cb + (long)k * VQ_D);
    float s = 0.0f;
    #pragma unroll 8
    for (int i = 0; i < VQ_D / 4; ++i) {
        float4 v = c4[i];
        s += v.x * v.x + v.y * v.y + v.z * v.z + v.w * v.w;
    }
    cnorms[k] = s;
    if (k == 0) loss_slot[0] = 0.0f;
}

// ---------------------------------------------------------------------------
// Kernel 3: WMMA distance + running argmin over one K-partition.
//   grid = 32 row-blocks x 8 K-partitions = 256 blocks, 256 thr (8 waves).
//   Each wave owns TWO 16-row A tiles (B fragment reuse -> 1:1 ds:wmma).
//   Codebook tiles arrive via double-buffered TENSOR_LOAD_TO_LDS.
// ---------------------------------------------------------------------------
__global__ __launch_bounds__(256) void vq_argmin_part(
    const unsigned short* __restrict__ ze_bf16,
    const unsigned short* __restrict__ cb_bf16,
    const float* __restrict__ cnorms,
    float* __restrict__ pdist,
    int*   __restrict__ pidx)
{
    __shared__ __bf16 smem[2][16 * LDS_STRIDE];

    const __bf16* ze_bf = (const __bf16*)ze_bf16;
    const __bf16* cb_bf = (const __bf16*)cb_bf16;

    const int lane    = threadIdx.x & 31;
    const int wave    = threadIdx.x >> 5;
    const int half    = lane >> 4;           // 0: lanes 0-15, 1: lanes 16-31
    const int col     = lane & 15;           // N-column / code-within-chunk
    const int part    = blockIdx.x & (VQ_PARTS - 1);
    const int rowBlk  = blockIdx.x >> 3;
    const int rowBase = rowBlk * 256 + wave * 32;
    const int kBase   = part * VQ_KPART;

    // Resident A fragments: 2 tiles x 8 K-slices (128 VGPRs).
    Frag32B A[2][8];
    #pragma unroll
    for (int t = 0; t < 2; ++t) {
        #pragma unroll
        for (int kd = 0; kd < 8; ++kd) {
            const __bf16* p = ze_bf + (long)(rowBase + t * 16 + col) * VQ_D
                                    + kd * 32 + half * 16;
            A[t][kd].u[0] = *(const v4u*)p;
            A[t][kd].u[1] = *(const v4u*)(p + 8);
        }
    }

    float best[2][8];
    int   bidx[2][8];
    #pragma unroll
    for (int t = 0; t < 2; ++t)
        #pragma unroll
        for (int r = 0; r < 8; ++r) { best[t][r] = 3.4e38f; bidx[t][r] = 0; }

    const unsigned lds0 = lds_offset(&smem[0][0]);
    const unsigned lds1 = lds_offset(&smem[1][0]);

    // prologue: DMA chunk 0 into buffer 0
    if (wave == 0)
        tdm_load_tile(cb_bf + (long)kBase * VQ_D, lds0);

    for (int chunk = 0; chunk < VQ_NCHUNK; ++chunk) {
        const int cur = chunk & 1;
        if (wave == 0) {
            if (chunk + 1 < VQ_NCHUNK) {
                // DMA next tile into the other buffer, then wait until only
                // that one is outstanding (TDM completes in order).
                tdm_load_tile(cb_bf + (long)(kBase + (chunk + 1) * 16) * VQ_D,
                              cur ? lds0 : lds1);
                __builtin_amdgcn_s_wait_tensorcnt((short)1);
            } else {
                __builtin_amdgcn_s_wait_tensorcnt((short)0);
            }
        }
        __syncthreads();                     // publish DMA'd tile to all waves

        const __bf16* sbase = &smem[cur][0];
        v8f acc0 = {};
        v8f acc1 = {};
        #pragma unroll
        for (int kd = 0; kd < 8; ++kd) {
            Frag32B Bf;
            const __bf16* s = sbase + col * LDS_STRIDE + kd * 32 + half * 16;
            Bf.u[0] = *(const v4u*)s;
            Bf.u[1] = *(const v4u*)(s + 8);
            acc0 = __builtin_amdgcn_wmma_f32_16x16x32_bf16(
                false, A[0][kd].bf, false, Bf.bf, (short)0, acc0, false, false);
            acc1 = __builtin_amdgcn_wmma_f32_16x16x32_bf16(
                false, A[1][kd].bf, false, Bf.bf, (short)0, acc1, false, false);
        }

        const int   code = kBase + chunk * 16 + col;
        const float cn   = cnorms[code];
        #pragma unroll
        for (int r = 0; r < 8; ++r) {
            float d0 = cn - 2.0f * acc0[r];          // ||x||^2 dropped
            float d1 = cn - 2.0f * acc1[r];
            if (d0 < best[0][r]) { best[0][r] = d0; bidx[0][r] = code; }
            if (d1 < best[1][r]) { best[1][r] = d1; bidx[1][r] = code; }
        }
        __syncthreads();                     // done reading buf before reuse
    }

    // reduce over the 16 lanes sharing each row; C layout:
    //   VGPR r: lanes 0-15 -> row base+r, lanes 16-31 -> row base+8+r
    #pragma unroll
    for (int t = 0; t < 2; ++t) {
        #pragma unroll
        for (int r = 0; r < 8; ++r) {
            float bd = best[t][r];
            int   bi = bidx[t][r];
            #pragma unroll
            for (int m = 8; m >= 1; m >>= 1) {
                float od = __shfl_xor(bd, m, 32);
                int   oi = __shfl_xor(bi, m, 32);
                if (od < bd || (od == bd && oi < bi)) { bd = od; bi = oi; }
            }
            if (col == 0) {
                const int row = rowBase + t * 16 + r + half * 8;
                pdist[part * VQ_N + row] = bd;
                pidx [part * VQ_N + row] = bi;
            }
        }
    }
}

// ---------------------------------------------------------------------------
// Kernel 4: merge the 8 per-partition argmin candidates per row.
// ---------------------------------------------------------------------------
__global__ __launch_bounds__(256) void vq_merge(
    const float* __restrict__ pdist, const int* __restrict__ pidx,
    int* __restrict__ idx_out, float* __restrict__ idxf_out)
{
    const int row = blockIdx.x * blockDim.x + threadIdx.x;  // < VQ_N
    float bd = pdist[row];
    int   bi = pidx[row];
    #pragma unroll
    for (int p = 1; p < VQ_PARTS; ++p) {
        float d = pdist[p * VQ_N + row];
        int   i = pidx [p * VQ_N + row];
        if (d < bd) { bd = d; bi = i; }   // ascending p => first-min tiebreak
    }
    idx_out[row]  = bi;
    idxf_out[row] = (float)bi;
}

// ---------------------------------------------------------------------------
// Kernel 5: gather z_q (fp32 codebook), write straight-through output and
// accumulate vq_loss = (1+BETA) * mean((z_e - z_q)^2).
// ---------------------------------------------------------------------------
__global__ __launch_bounds__(256) void vq_finalize(
    const float* __restrict__ ze, const float* __restrict__ cb,
    const int* __restrict__ idx, float* __restrict__ zq,
    float* __restrict__ loss)
{
    const int t   = blockIdx.x * blockDim.x + threadIdx.x;  // < VQ_ND
    const int row = t >> 8;
    const int c   = t & 255;
    const int k   = idx[row];
    const float q = cb[(long)k * VQ_D + c];
    zq[t] = q;                                  // forward value of z_q_st
    float d = ze[t] - q;
    float e = d * d;

    #pragma unroll
    for (int m = 16; m >= 1; m >>= 1) e += __shfl_xor(e, m, 32);

    __shared__ float red[8];
    const int lane = threadIdx.x & 31;
    const int wv   = threadIdx.x >> 5;
    if (lane == 0) red[wv] = e;
    __syncthreads();
    if (threadIdx.x == 0) {
        float s = 0.0f;
        #pragma unroll
        for (int i = 0; i < 8; ++i) s += red[i];
        atomicAdd(loss, s * ((1.0f + VQ_BETA) / (float)VQ_ND));
    }
}

// ---------------------------------------------------------------------------
// Host launcher.
//   d_in[0] = z_e [4,2048,256] f32      d_in[1] = codebook [8192,256] f32
//   d_out   = z_q_st (2,097,152 f32) | indices (8192 f32) | vq_loss (1 f32)
//   d_ws    : cb_bf16 4MB | ze_bf16 4MB | idx 32KB | cnorms 32KB
//           | pdist 256KB | pidx 256KB          (total ~8.6 MB)
// ---------------------------------------------------------------------------
extern "C" void kernel_launch(void* const* d_in, const int* in_sizes, int n_in,
                              void* d_out, int out_size, void* d_ws, size_t ws_size,
                              hipStream_t stream)
{
    const float* z_e = (const float*)d_in[0];
    const float* cb  = (const float*)d_in[1];

    float* out  = (float*)d_out;
    float* zq   = out;                    // 2,097,152
    float* idxf = out + VQ_ND;            // 8,192
    float* loss = out + VQ_ND + VQ_N;     // 1

    char* ws = (char*)d_ws;
    unsigned short* cb_bf  = (unsigned short*)(ws);                   // 4,194,304 B
    unsigned short* ze_bf  = (unsigned short*)(ws + 4194304);         // 4,194,304 B
    int*            idx_i  = (int*)          (ws + 8388608);          //    32,768 B
    float*          cnorms = (float*)        (ws + 8421376);          //    32,768 B
    float*          pdist  = (float*)        (ws + 8454144);          //   262,144 B
    int*            pidx   = (int*)          (ws + 8716288);          //   262,144 B

    vq_prep_convert<<<VQ_ND / 4 / 256,      256, 0, stream>>>(z_e, cb, ze_bf, cb_bf);
    vq_prep_cnorms <<<VQ_K / 256,           256, 0, stream>>>(cb, cnorms, loss);
    vq_argmin_part <<<32 * VQ_PARTS,        256, 0, stream>>>(ze_bf, cb_bf, cnorms, pdist, pidx);
    vq_merge       <<<VQ_N / 256,           256, 0, stream>>>(pdist, pidx, idx_i, idxf);
    vq_finalize    <<<VQ_ND / 256,          256, 0, stream>>>(z_e, cb, idx_i, zq, loss);
}
